// Llama_84894323573351
// MI455X (gfx1250) — matrix operand order repaired and checked by reading, not compiled, not run on hardware
//
#include <hip/hip_runtime.h>
#include <hip/hip_bf16.h>

// ---------------------------------------------------------------------------
// MI455X (gfx1250) Llama forward. Strategy: all GEMMs through
// v_wmma_f32_16x16x32_bf16 (wave32, 16x16 tiles, K=32/issue, f32 accum).
// Weights are re-packed once per call into a contiguous-k (k = 16*(lane/16) + e) B-operand lane
// layout (same k order as the A fragments) as bf16 so B fragments are single 32B per-lane loads.
// Activations stay f32 and are converted to bf16 in the A-fragment load.
// Wave tile is 32x64: 2 A-frags x 4 B-frags -> 8 WMMAs per K=32 step, so
// every B fragment is reused twice (halves B traffic per FLOP).
// ---------------------------------------------------------------------------

typedef __attribute__((ext_vector_type(16))) __bf16 v16bf;
typedef __attribute__((ext_vector_type(8)))  float  v8f;

#define D_MODEL 256
#define SEQ     1024
#define BATCHN  2
#define HEADS   8
#define VOCAB   32000
#define NLAYERS 4
#define BS      (BATCHN * SEQ)          // 2048 rows of activations

// ---------------------------------------------------------------------------
// Weight pre-pack: W[K][N] f32 row-major  ->  packed bf16 in WMMA B layout.
// Packed index = (((nt*KT + kt)*32 + lane)*16 + e)
//   maps to W[ kt*32 + (lane>=16 ? 16 : 0) + e ][ nt*16 + (lane&15) ]
// blockIdx.z selects matrix (e.g. attention head); stride = K*N elements.
// ---------------------------------------------------------------------------
__global__ __launch_bounds__(256)
void k_pack(const float* __restrict__ W, __bf16* __restrict__ P, int K, int N) {
  const size_t base = (size_t)blockIdx.z * (size_t)K * (size_t)N;
  const int idx = blockIdx.x * 256 + threadIdx.x;
  if (idx >= K * N) return;
  const int KT   = K >> 5;
  const int e    = idx & 15;
  const int lane = (idx >> 4) & 31;
  const int tl   = idx >> 9;            // linear tile index = nt*KT + kt
  const int kt   = tl % KT;
  const int nt   = tl / KT;
  const int k = kt * 32 + ((lane >> 4) << 4) + e;
  const int n = nt * 16 + (lane & 15);
  P[base + (size_t)idx] = (__bf16)W[base + (size_t)k * N + n];
}

// A fragment (16-bit A 16x32): lane<16 -> K {0..15},
// lane>=16 -> K {16..31}; converted f32 -> bf16 in registers.
__device__ __forceinline__ v16bf load_a_frag(const float* __restrict__ A,
                                             int lda, int mt, int kt,
                                             int m, int hl) {
  const float* arow = A + (size_t)(mt * 16 + m) * lda + kt * 32 + hl * 16;
  v16bf a;
#pragma unroll
  for (int e = 0; e < 8; ++e) a[e] = (__bf16)arow[e];
#pragma unroll
  for (int e = 0; e < 8; ++e) a[8 + e] = (__bf16)arow[8 + e];
  return a;
}

// ---------------------------------------------------------------------------
// Generic WMMA GEMM: C[M,N] = scale * (A[M,K] x B) + bias (+C if accum).
// One wave -> 32x64 tile; block of 128 threads (4 waves) -> 32x256 per block.
//   BMODE 0: B = packed bf16 weights (per-lane contiguous 32B loads)
//   BMODE 1: B = f32 row-major source used TRANSPOSED (scores: B[k][n]=Q[n][k])
//   BMODE 2: B = f32 row-major source direct (A.V: strided column gathers)
// blockIdx.z strides B (bytes), C and bias (elements) for per-head batching.
// Control flow is wave-uniform so EXEC is all-ones at every WMMA (ISA 7.12).
// ---------------------------------------------------------------------------
template <int BMODE>
__global__ __launch_bounds__(128)
void k_gemm(const float* __restrict__ A, int lda,
            const void* __restrict__ Bv, int ldb,
            const float* __restrict__ bias,
            float scale, int accum,
            float* __restrict__ C, int ldc, int K,
            unsigned long long bStrideBytes,
            unsigned long long cStrideZ,
            unsigned long long biasStrideZ) {
  const int lane   = threadIdx.x & 31;
  const int wave   = threadIdx.x >> 5;
  const int ntBase = blockIdx.x * 16 + wave * 4;   // 4 n-tiles of 16 per wave
  const int mt0    = blockIdx.y * 2;               // 2 m-tiles of 16 per wave
  const int KT     = K >> 5;

  const char*  Bz    = (const char*)Bv + (size_t)blockIdx.z * bStrideBytes;
  float*       Cz    = C + (size_t)blockIdx.z * cStrideZ;
  const float* biasZ = bias ? bias + (size_t)blockIdx.z * biasStrideZ : nullptr;

  const int m  = lane & 15;    // row within 16 (A) / col within 16 (B)
  const int hl = lane >> 4;    // lane-group

  v8f acc[2][4];
  v8f zero = {};
#pragma unroll
  for (int mi = 0; mi < 2; ++mi)
#pragma unroll
    for (int t = 0; t < 4; ++t) acc[mi][t] = zero;

  for (int kt = 0; kt < KT; ++kt) {
    const v16bf a0 = load_a_frag(A, lda, mt0 + 0, kt, m, hl);
    const v16bf a1 = load_a_frag(A, lda, mt0 + 1, kt, m, hl);

#pragma unroll
    for (int t = 0; t < 4; ++t) {
      const int nt = ntBase + t;
      v16bf b;
      if (BMODE == 0) {
        const __bf16* bp =
            (const __bf16*)Bz + (((size_t)nt * KT + kt) * 32 + lane) * 16;
        b = *(const v16bf*)bp;
        // Speculative prefetch of the next K-slice; past-the-end addresses
        // are silently dropped (ISA 10.5), so no guard branch needed.
        __builtin_prefetch(bp + (size_t)32 * 16, 0, 1);
      } else if (BMODE == 1) {
        // B[k][n] = Bsrc[n][k]: per-lane contiguous 16 f32 from row n.
        const float* brow =
            (const float*)Bz + (size_t)(nt * 16 + m) * ldb + kt * 32 + hl * 16;
#pragma unroll
        for (int e = 0; e < 16; ++e) b[e] = (__bf16)brow[e];
      } else {
        // Row-major B: column n gathered with stride ldb.
        const float* bp =
            (const float*)Bz + (size_t)(kt * 32 + hl * 16) * ldb + nt * 16 + m;
#pragma unroll
        for (int e = 0; e < 16; ++e) b[e] = (__bf16)bp[(size_t)e * ldb];
      }
      acc[0][t] = __builtin_amdgcn_wmma_f32_16x16x32_bf16(
          false, a0, false, b, (short)0, acc[0][t], false, false);
      acc[1][t] = __builtin_amdgcn_wmma_f32_16x16x32_bf16(
          false, a1, false, b, (short)0, acc[1][t], false, false);
    }
  }

  // C/D layout: VGPR r, lane L -> row 16*mt + (L>=16?8:0)+r, col = L&15.
#pragma unroll
  for (int mi = 0; mi < 2; ++mi) {
    const int mrow = (mt0 + mi) * 16 + hl * 8;
#pragma unroll
    for (int t = 0; t < 4; ++t) {
      const int n = (ntBase + t) * 16 + m;
      const float bb = biasZ ? biasZ[n] : 0.0f;
#pragma unroll
      for (int r = 0; r < 8; ++r) {
        const size_t idx = (size_t)(mrow + r) * ldc + n;
        float v = acc[mi][t][r] * scale + bb;
        if (accum) v += Cz[idx];
        Cz[idx] = v;
      }
    }
  }
}

// ---------------------------------------------------------------------------
// Elementwise / reduction kernels
// ---------------------------------------------------------------------------
__global__ __launch_bounds__(256)
void k_embed(const int* __restrict__ tok, const float* __restrict__ E,
             float* __restrict__ X, int total) {
  const int idx = blockIdx.x * 256 + threadIdx.x;
  if (idx >= total) return;
  const int d  = idx & (D_MODEL - 1);
  const int bs = idx >> 8;
  X[idx] = E[(size_t)tok[bs] * D_MODEL + d];
}

__global__ __launch_bounds__(256)
void k_rmsnorm(const float* __restrict__ X, const float* __restrict__ g,
               float* __restrict__ Y) {
  __shared__ float sh[8];
  __shared__ float inv;
  const int row = blockIdx.x, t = threadIdx.x;
  const float v = X[(size_t)row * D_MODEL + t];
  float s = v * v;
#pragma unroll
  for (int o = 16; o; o >>= 1) s += __shfl_xor(s, o, 32);
  if ((t & 31) == 0) sh[t >> 5] = s;
  __syncthreads();
  if (t == 0) {
    float tot = 0.0f;
#pragma unroll
    for (int i = 0; i < 8; ++i) tot += sh[i];
    inv = rsqrtf(tot * (1.0f / D_MODEL) + 1e-6f);
  }
  __syncthreads();
  Y[(size_t)row * D_MODEL + t] = v * inv * g[t];
}

// RoPE in place over [H,B,S,D]; pair p uses theta = 10000^(-2(p-1)/D).
__global__ __launch_bounds__(256)
void k_rope(float* __restrict__ T, int totalPairs) {
  const int idx = blockIdx.x * 256 + threadIdx.x;
  if (idx >= totalPairs) return;
  const int p = idx & 127;
  const int s = (idx >> 7) & (SEQ - 1);
  const float theta =
      __expf(-2.0f * ((float)p - 1.0f) * (1.0f / D_MODEL) * 9.210340371976184f);
  float sn, cs;
  __sincosf((float)s * theta, &sn, &cs);
  float* base = T + ((size_t)(idx >> 7)) * D_MODEL + 2 * p;
  const float xe = base[0], xo = base[1];
  base[0] =  xe * cs + xo * sn;
  base[1] = -xe * sn + xo * cs;
}

// Causal softmax over row i of P[S,S]; mask j>i with -1e9 like the reference.
__global__ __launch_bounds__(256)
void k_softmax(float* __restrict__ P, int S) {
  __shared__ float sh[8];
  __shared__ float bval;
  const int i = blockIdx.x, t = threadIdx.x;
  float* row = P + (size_t)i * S;
  float mx = -3.0e38f;
  for (int j = t; j < S; j += 256) {
    const float v = (j <= i) ? row[j] : -1.0e9f;
    mx = fmaxf(mx, v);
  }
#pragma unroll
  for (int o = 16; o; o >>= 1) mx = fmaxf(mx, __shfl_xor(mx, o, 32));
  if ((t & 31) == 0) sh[t >> 5] = mx;
  __syncthreads();
  if (t == 0) {
    float m = sh[0];
    for (int w = 1; w < 8; ++w) m = fmaxf(m, sh[w]);
    bval = m;
  }
  __syncthreads();
  mx = bval;
  __syncthreads();
  float sum = 0.0f;
  for (int j = t; j < S; j += 256) {
    const float v = (j <= i) ? row[j] : -1.0e9f;
    const float e = __expf(v - mx);
    row[j] = e;
    sum += e;
  }
#pragma unroll
  for (int o = 16; o; o >>= 1) sum += __shfl_xor(sum, o, 32);
  if ((t & 31) == 0) sh[t >> 5] = sum;
  __syncthreads();
  if (t == 0) {
    float m = 0.0f;
    for (int w = 0; w < 8; ++w) m += sh[w];
    bval = 1.0f / m;
  }
  __syncthreads();
  const float r = bval;
  for (int j = t; j < S; j += 256) row[j] *= r;
}

// concat heads: CC[b*S+s][h*D+d] = O[h][b][s][d]
__global__ __launch_bounds__(256)
void k_concat(const float* __restrict__ O, float* __restrict__ CC, int total) {
  const int idx = blockIdx.x * 256 + threadIdx.x;
  if (idx >= total) return;
  const int col = idx & (HEADS * D_MODEL - 1);
  const int bs  = idx >> 11;
  const int h = col >> 8, d = col & (D_MODEL - 1);
  const int b = bs >> 10, s = bs & (SEQ - 1);
  CC[idx] = O[(((size_t)(h * BATCHN + b) * SEQ + s)) * D_MODEL + d];
}

// Y (+)= g*sigmoid(g)*l
__global__ __launch_bounds__(256)
void k_swiglu(const float* __restrict__ G, const float* __restrict__ L,
              float* __restrict__ Y, int n, int add) {
  const int i = blockIdx.x * 256 + threadIdx.x;
  if (i >= n) return;
  const float g = G[i];
  const float v = g * (1.0f / (1.0f + __expf(-g))) * L[i];
  Y[i] = add ? Y[i] + v : v;
}

// ---------------------------------------------------------------------------
// Host orchestration
// ---------------------------------------------------------------------------
extern "C" void kernel_launch(void* const* d_in, const int* in_sizes, int n_in,
                              void* d_out, int out_size, void* d_ws,
                              size_t ws_size, hipStream_t stream) {
  (void)in_sizes; (void)n_in; (void)out_size; (void)ws_size;

  const int* tokens = (const int*)d_in[0];
  const float* embed = (const float*)d_in[1];
  auto LP = [&](int l, int j) { return (const float*)d_in[2 + l * 16 + j]; };
  // layer keys: 0 rms1,1 Wq,2 bq,3 Wk,4 bk,5 Wv,6 bv,7 Wo,8 bo,9 rms2,
  //             10 ff_W,11 ff_b,12 sg_Wg,13 sg_bg,14 sg_Wl,15 sg_bl
  const float* f_W     = (const float*)d_in[66];
  const float* f_b     = (const float*)d_in[67];
  const float* f_sg_Wg = (const float*)d_in[68];
  const float* f_sg_bg = (const float*)d_in[69];
  const float* f_sg_Wl = (const float*)d_in[70];
  const float* f_sg_bl = (const float*)d_in[71];
  const float* out_W   = (const float*)d_in[72];
  const float* out_b   = (const float*)d_in[73];

  // ---- workspace carve-up (~100 MB total) ----
  float* ws = (float*)d_ws;
  size_t off = 0;
  auto falloc = [&](size_t n) { float* p = ws + off; off += n; return p; };
  float* X  = falloc((size_t)BS * D_MODEL);          // running activations
  float* XN = falloc((size_t)BS * D_MODEL);          // norm / residual branch
  float* HB = falloc((size_t)BS * D_MODEL);          // ffn hidden
  float* GB = falloc((size_t)BS * D_MODEL);          // swiglu gate
  float* LB = falloc((size_t)BS * D_MODEL);          // swiglu linear
  float* Q  = falloc((size_t)HEADS * BS * D_MODEL);  // Q, later per-head O
  float* Kb = falloc((size_t)HEADS * BS * D_MODEL);  // K, later concat buffer
  float* Vb = falloc((size_t)HEADS * BS * D_MODEL);  // V
  float* SC = falloc((size_t)SEQ * SEQ);             // scores, reused per (h,b)

  __bf16* pkbase = (__bf16*)(ws + off);
  size_t pkoff = 0;
  auto palloc = [&](size_t n) { __bf16* p = pkbase + pkoff; pkoff += n; return p; };
  __bf16 *pWq[NLAYERS], *pWk[NLAYERS], *pWv[NLAYERS], *pWo[NLAYERS];
  __bf16 *pFF[NLAYERS], *pSg[NLAYERS], *pSl[NLAYERS];
  for (int l = 0; l < NLAYERS; ++l) {
    pWq[l] = palloc((size_t)HEADS * D_MODEL * D_MODEL);
    pWk[l] = palloc((size_t)HEADS * D_MODEL * D_MODEL);
    pWv[l] = palloc((size_t)HEADS * D_MODEL * D_MODEL);
    pWo[l] = palloc((size_t)HEADS * D_MODEL * D_MODEL);
    pFF[l] = palloc((size_t)D_MODEL * D_MODEL);
    pSg[l] = palloc((size_t)D_MODEL * D_MODEL);
    pSl[l] = palloc((size_t)D_MODEL * D_MODEL);
  }
  __bf16* pfW  = palloc((size_t)D_MODEL * D_MODEL);
  __bf16* pfg  = palloc((size_t)D_MODEL * D_MODEL);
  __bf16* pfl  = palloc((size_t)D_MODEL * D_MODEL);
  __bf16* pOut = palloc((size_t)D_MODEL * VOCAB);

  auto pack = [&](const float* W, __bf16* P, int K, int N, int nz) {
    dim3 g((unsigned)((K * N + 255) / 256), 1, (unsigned)nz);
    hipLaunchKernelGGL(k_pack, g, dim3(256), 0, stream, W, P, K, N);
  };
  auto gemm = [&](int mode, const float* A, int lda, const void* Bv, int ldb,
                  const float* bias, float scale, int accum, float* C, int ldc,
                  int M, int N, int Kd, int nz, unsigned long long bStrB,
                  unsigned long long cStr, unsigned long long biasStr) {
    dim3 g((unsigned)(N / 256), (unsigned)(M / 32), (unsigned)nz);
    dim3 blk(128);
    if (mode == 0)
      hipLaunchKernelGGL((k_gemm<0>), g, blk, 0, stream, A, lda, Bv, ldb, bias,
                         scale, accum, C, ldc, Kd, bStrB, cStr, biasStr);
    else if (mode == 1)
      hipLaunchKernelGGL((k_gemm<1>), g, blk, 0, stream, A, lda, Bv, ldb, bias,
                         scale, accum, C, ldc, Kd, bStrB, cStr, biasStr);
    else
      hipLaunchKernelGGL((k_gemm<2>), g, blk, 0, stream, A, lda, Bv, ldb, bias,
                         scale, accum, C, ldc, Kd, bStrB, cStr, biasStr);
  };

  // ---- one-time weight repack (runs every call; deterministic) ----
  for (int l = 0; l < NLAYERS; ++l) {
    pack(LP(l, 1),  pWq[l], D_MODEL, D_MODEL, HEADS);
    pack(LP(l, 3),  pWk[l], D_MODEL, D_MODEL, HEADS);
    pack(LP(l, 5),  pWv[l], D_MODEL, D_MODEL, HEADS);
    pack(LP(l, 7),  pWo[l], HEADS * D_MODEL, D_MODEL, 1);
    pack(LP(l, 10), pFF[l], D_MODEL, D_MODEL, 1);
    pack(LP(l, 12), pSg[l], D_MODEL, D_MODEL, 1);
    pack(LP(l, 14), pSl[l], D_MODEL, D_MODEL, 1);
  }
  pack(f_W, pfW, D_MODEL, D_MODEL, 1);
  pack(f_sg_Wg, pfg, D_MODEL, D_MODEL, 1);
  pack(f_sg_Wl, pfl, D_MODEL, D_MODEL, 1);
  pack(out_W, pOut, D_MODEL, VOCAB, 1);

  const unsigned long long headPkBytes = (unsigned long long)D_MODEL * D_MODEL * 2;
  const unsigned long long headAct     = (unsigned long long)BS * D_MODEL;

  // ---- embedding ----
  {
    const int total = BS * D_MODEL;
    hipLaunchKernelGGL(k_embed, dim3((total + 255) / 256), dim3(256), 0, stream,
                       tokens, embed, X, total);
  }

  // ---- transformer blocks ----
  for (int l = 0; l < NLAYERS; ++l) {
    // XN = rmsnorm(X, rms1)
    hipLaunchKernelGGL(k_rmsnorm, dim3(BS), dim3(D_MODEL), 0, stream, X, LP(l, 0), XN);

    // QKV: per head, [2048x256] x packed[256x256] (+bias); z = head
    gemm(0, XN, D_MODEL, pWq[l], 0, LP(l, 2), 1.0f, 0, Q,  D_MODEL,
         BS, D_MODEL, D_MODEL, HEADS, headPkBytes, headAct, D_MODEL);
    gemm(0, XN, D_MODEL, pWk[l], 0, LP(l, 4), 1.0f, 0, Kb, D_MODEL,
         BS, D_MODEL, D_MODEL, HEADS, headPkBytes, headAct, D_MODEL);
    gemm(0, XN, D_MODEL, pWv[l], 0, LP(l, 6), 1.0f, 0, Vb, D_MODEL,
         BS, D_MODEL, D_MODEL, HEADS, headPkBytes, headAct, D_MODEL);

    // RoPE on q, k, v
    const int pairs = HEADS * BS * (D_MODEL / 2);
    hipLaunchKernelGGL(k_rope, dim3((pairs + 255) / 256), dim3(256), 0, stream, Q,  pairs);
    hipLaunchKernelGGL(k_rope, dim3((pairs + 255) / 256), dim3(256), 0, stream, Kb, pairs);
    hipLaunchKernelGGL(k_rope, dim3((pairs + 255) / 256), dim3(256), 0, stream, Vb, pairs);

    // attention, streamed per (head, batch); scores buffer reused
    for (int h = 0; h < HEADS; ++h) {
      for (int b = 0; b < BATCHN; ++b) {
        const size_t slice = (size_t)(h * BATCHN + b) * SEQ * D_MODEL;
        // scores[i,j] = (K[i,:] . Q[j,:]) / 16  -> BMODE 1 (Q as B^T source)
        gemm(1, Kb + slice, D_MODEL, Q + slice, D_MODEL, nullptr, 0.0625f, 0,
             SC, SEQ, SEQ, SEQ, D_MODEL, 1, 0, 0, 0);
        hipLaunchKernelGGL(k_softmax, dim3(SEQ), dim3(256), 0, stream, SC, SEQ);
        // o = P x V -> overwrite Q slice (dead after its scores GEMM)
        gemm(2, SC, SEQ, Vb + slice, D_MODEL, nullptr, 1.0f, 0,
             Q + slice, D_MODEL, SEQ, D_MODEL, SEQ, 1, 0, 0, 0);
      }
    }

    // concat heads into Kb (same footprint, K is dead now)
    {
      const int total = BS * HEADS * D_MODEL;
      hipLaunchKernelGGL(k_concat, dim3((total + 255) / 256), dim3(256), 0,
                         stream, Q, Kb, total);
    }
    // XN += concat @ Wo + bo   (residual base is the rms1 output)
    gemm(0, Kb, HEADS * D_MODEL, pWo[l], 0, LP(l, 8), 1.0f, 1, XN, D_MODEL,
         BS, D_MODEL, HEADS * D_MODEL, 1, 0, 0, 0);

    // X = rmsnorm(XN, rms2)
    hipLaunchKernelGGL(k_rmsnorm, dim3(BS), dim3(D_MODEL), 0, stream, XN, LP(l, 9), X);

    // ffn: h = X@ff_W+ff_b ; X += swiglu(h)
    gemm(0, X,  D_MODEL, pFF[l], 0, LP(l, 11), 1.0f, 0, HB, D_MODEL,
         BS, D_MODEL, D_MODEL, 1, 0, 0, 0);
    gemm(0, HB, D_MODEL, pSg[l], 0, LP(l, 13), 1.0f, 0, GB, D_MODEL,
         BS, D_MODEL, D_MODEL, 1, 0, 0, 0);
    gemm(0, HB, D_MODEL, pSl[l], 0, LP(l, 15), 1.0f, 0, LB, D_MODEL,
         BS, D_MODEL, D_MODEL, 1, 0, 0, 0);
    {
      const int total = BS * D_MODEL;
      hipLaunchKernelGGL(k_swiglu, dim3((total + 255) / 256), dim3(256), 0,
                         stream, GB, LB, X, total, 1);
    }
  }

  // ---- final head ----
  gemm(0, X,  D_MODEL, pfW, 0, f_b,     1.0f, 0, HB, D_MODEL,
       BS, D_MODEL, D_MODEL, 1, 0, 0, 0);
  gemm(0, HB, D_MODEL, pfg, 0, f_sg_bg, 1.0f, 0, GB, D_MODEL,
       BS, D_MODEL, D_MODEL, 1, 0, 0, 0);
  gemm(0, HB, D_MODEL, pfl, 0, f_sg_bl, 1.0f, 0, LB, D_MODEL,
       BS, D_MODEL, D_MODEL, 1, 0, 0, 0);
  {
    const int total = BS * D_MODEL;
    hipLaunchKernelGGL(k_swiglu, dim3((total + 255) / 256), dim3(256), 0,
                       stream, GB, LB, XN, total, 0);   // XN = f
  }
  // logits = f @ out_W + out_b  -> d_out [2048 x 32000] f32
  gemm(0, XN, D_MODEL, pOut, 0, out_b, 1.0f, 0, (float*)d_out, VOCAB,
       BS, VOCAB, D_MODEL, 1, 0, 0, 0);
}
